// ScaledDotProductAttention_9517647528307
// MI455X (gfx1250) — matrix operand order repaired and checked by reading, not compiled, not run on hardware
//
#include <hip/hip_runtime.h>
#include <hip/hip_bf16.h>

typedef __attribute__((ext_vector_type(16))) _Float16 v16h;
typedef __attribute__((ext_vector_type(8)))  float    v8f;

namespace {
constexpr int   kB      = 32;
constexpr int   kN      = 2048;
constexpr int   kD      = 64;
constexpr int   kQT     = 16;          // query rows per workgroup
constexpr int   kWaves  = 8;           // 256 threads / wave32
constexpr int   kTiles  = kN / 16;     // 128 key tiles
constexpr int   kStride = kN + 4;      // padded LDS row stride (floats)
constexpr float kScale  = 0.125f;      // 1/sqrt(64)
constexpr float kMinVal = -1e9f;
}

__global__ __launch_bounds__(256, 1)
void sdpa_fwd(const float* __restrict__ q, const float* __restrict__ k,
              const float* __restrict__ v, const int* __restrict__ mask,
              float* __restrict__ out, float* __restrict__ attn)
{
  __shared__ float sS[kQT][kStride];   // scores -> probs (~128KB)
  __shared__ float sRed[kQT][16];      // row reductions
  __shared__ float sRow[kQT];          // row max / inv-sum
  __shared__ float sOut[kQT][kD];      // output accumulator

  const int b    = blockIdx.y;
  const int q0   = blockIdx.x * kQT;
  const int tid  = threadIdx.x;
  const int wave = tid >> 5;
  const int lane = tid & 31;
  const int g    = lane >> 4;          // half-wave id (0/1)
  const int ln   = lane & 15;

  // zero output accumulator (visible after first barrier)
  for (int i = tid; i < kQT * kD; i += 256) ((float*)sOut)[i] = 0.0f;

  // ---- Q tile A-fragments: 16x32 f16 layout (two fragments cover d=0..63) ----
  // A fragment, contiguous k order used here (same as B): lane M = lane%16; half h -> K = h + 16*g
  const float* qrow = q + ((size_t)b * kN + q0 + ln) * kD;
  v16h aq[2];
  #pragma unroll
  for (int f = 0; f < 2; ++f)
    #pragma unroll
    for (int h = 0; h < 16; ++h) {
      const int d = f * 32 + h + 16 * g;
      aq[f][h] = (_Float16)qrow[d];
    }

  // ---- Phase 1: S = (Q K^T) * scale, masked; stage rows in LDS ----
  for (int kt = wave; kt < kTiles; kt += kWaves) {
    const int k0 = kt * 16;
    // B fragment (32x16), same contiguous k order as A: lane N = lane%16; half h -> K = h + 16*g
    // B[K][N] = Kmat[k0+N][f*32 + K]  (16 consecutive floats per lane)
    const float* krow = k + ((size_t)b * kN + k0 + ln) * kD + 16 * g;
    v8f c = {};
    #pragma unroll
    for (int f = 0; f < 2; ++f) {
      const float4* k4 = (const float4*)(krow + f * 32);
      v16h bk;
      #pragma unroll
      for (int j = 0; j < 4; ++j) {
        const float4 t4 = k4[j];
        bk[4 * j + 0] = (_Float16)t4.x;
        bk[4 * j + 1] = (_Float16)t4.y;
        bk[4 * j + 2] = (_Float16)t4.z;
        bk[4 * j + 3] = (_Float16)t4.w;
      }
      c = __builtin_amdgcn_wmma_f32_16x16x32_f16(false, aq[f], false, bk,
                                                 (short)0, c, false, false);
    }
    // C layout: VGPR r -> M = r + 8*g, N = ln
    // Mask is a 536MB single-use stream: load non-temporal so it does not
    // evict the L2-resident K/V working set (1MB/batch, reused by 128 WGs).
    const int* mptr = mask + ((size_t)b * kN + q0) * kN + (size_t)(k0 + ln);
    #pragma unroll
    for (int r = 0; r < 8; ++r) {
      const int m  = r + 8 * g;
      const int mv = __builtin_nontemporal_load(mptr + (size_t)m * kN);
      sS[m][k0 + ln] = (mv > 0) ? c[r] * kScale : kMinVal;
    }
  }
  __syncthreads();

  // ---- Phase 2: row softmax (16 threads per row), write attn to global ----
  const int row = tid >> 4;
  const int col = tid & 15;

  float mx = -__builtin_inff();
  for (int j = 0; j < kN / 16; ++j) mx = fmaxf(mx, sS[row][col + 16 * j]);
  sRed[row][col] = mx;
  __syncthreads();
  if (col == 0) {
    float m2 = -__builtin_inff();
    #pragma unroll
    for (int i = 0; i < 16; ++i) m2 = fmaxf(m2, sRed[row][i]);
    sRow[row] = m2;
  }
  __syncthreads();
  const float rmax = sRow[row];

  float sum = 0.0f;
  for (int j = 0; j < kN / 16; ++j) {
    const int c2 = col + 16 * j;
    const float e = __expf(sS[row][c2] - rmax);
    sS[row][c2] = e;
    sum += e;
  }
  sRed[row][col] = sum;
  __syncthreads();
  if (col == 0) {
    float s2 = 0.0f;
    #pragma unroll
    for (int i = 0; i < 16; ++i) s2 += sRed[row][i];
    sRow[row] = 1.0f / s2;
  }
  __syncthreads();
  const float rinv = sRow[row];

  // attn is a 536MB single-use stream: write non-temporal (NT hint keeps it
  // out of the far cache so K/V stay resident for the sibling workgroups).
  float* arow = attn + ((size_t)b * kN + q0 + row) * kN;
  for (int j = 0; j < kN / 16; ++j) {
    const int c2 = col + 16 * j;
    const float p = sS[row][c2] * rinv;
    sS[row][c2] = p;
    __builtin_nontemporal_store(p, arow + c2);  // coalesced: 16 lanes/row
  }
  __syncthreads();

  // ---- Phase 3: O = P * V ; each wave owns a 256-deep K slice ----
  v8f acc[4] = {};
  #pragma unroll
  for (int ch = 0; ch < kN / (32 * kWaves); ++ch) {   // 8 chunks of 32
    const int kb = wave * (kN / kWaves) + ch * 32;
    // A fragment from probs in LDS
    v16h ap;
    #pragma unroll
    for (int h = 0; h < 16; ++h) {
      const int kk = kb + h + 16 * g;
      ap[h] = (_Float16)sS[ln][kk];
    }
    #pragma unroll
    for (int nt = 0; nt < 4; ++nt) {
      // B fragment from V: B[K][Ncol] = V[kb+K][nt*16+Ncol], K = h + 16*g
      const float* vbase = v + ((size_t)b * kN + kb + 16 * g) * kD + nt * 16 + ln;
      v16h bv;
      #pragma unroll
      for (int h = 0; h < 16; ++h)
        bv[h] = (_Float16)vbase[(size_t)h * kD];
      acc[nt] = __builtin_amdgcn_wmma_f32_16x16x32_f16(false, ap, false, bv,
                                                       (short)0, acc[nt], false, false);
    }
  }
  // cross-wave reduction via LDS float atomics (ds_add_f32)
  #pragma unroll
  for (int nt = 0; nt < 4; ++nt)
    #pragma unroll
    for (int r = 0; r < 8; ++r)
      atomicAdd(&sOut[r + 8 * g][nt * 16 + ln], acc[nt][r]);
  __syncthreads();

  for (int i = tid; i < kQT * kD; i += 256) {
    const int m = i / kD, d = i % kD;
    __builtin_nontemporal_store(((const float*)sOut)[i],
                                out + ((size_t)b * kN + q0 + m) * kD + d);
  }
}

extern "C" void kernel_launch(void* const* d_in, const int* in_sizes, int n_in,
                              void* d_out, int out_size, void* d_ws, size_t ws_size,
                              hipStream_t stream) {
  const float* q    = (const float*)d_in[0];
  const float* k    = (const float*)d_in[1];
  const float* v    = (const float*)d_in[2];
  const int*   mask = (const int*)d_in[3];
  float* out  = (float*)d_out;
  float* attn = out + (size_t)kB * kN * kD;   // outputs concatenated: [output, attn]

  dim3 grid(kN / kQT, kB);
  sdpa_fwd<<<grid, 256, 0, stream>>>(q, k, v, mask, out, attn);
}
